// LSTM_1666447311077
// MI455X (gfx1250) — matrix operand order repaired and checked by reading, not compiled, not run on hardware
//
#include <hip/hip_runtime.h>
#include <hip/hip_bf16.h>

// ---------------------------------------------------------------------------
// 2-layer LSTM (B=256,T=1024,IN=64,H=128) + FC head, fused persistent kernel.
// 16 WGs x 256 threads (8 waves, wave32). Each WG owns 16 batch rows for the
// whole sequence; each wave owns 16 h-columns (== 4 gate tiles i/f/g/o).
// W_hh0/W_hh1 live in LDS (256KB of the 320KB/WGP); per-wave W_ih0/W_ih1/W_fc
// slices live in registers. GEMMs via v_wmma_f32_16x16x32_f16, f32 cell state
// in registers for all 1024 steps.
// ---------------------------------------------------------------------------

typedef _Float16 h8   __attribute__((ext_vector_type(8)));
typedef _Float16 v16h __attribute__((ext_vector_type(16)));
typedef float    v8f  __attribute__((ext_vector_type(8)));

#define IN_   64
#define H_    128
#define OUT_  64
#define B_    256
#define T_    1024

// f16 weight workspace layout (element offsets)
#define OFF_WHH0  0                         // 512*128
#define OFF_WIH0  (OFF_WHH0 + 512*128)      // 512*64
#define OFF_WHH1  (OFF_WIH0 + 512*64)       // 512*128
#define OFF_WIH1  (OFF_WHH1 + 512*128)      // 512*128
#define OFF_WFC   (OFF_WIH1 + 512*128)      // 64*128
#define F16_TOTAL (OFF_WFC  + 64*128)       // 237568 halfs
#define BIAS_BYTE_OFF (F16_TOTAL * 2)       // then 1024 f32 fused biases

// dynamic-LDS partition (half-element offsets)
#define SM_WHH0   0                         // 512*128 = 65536
#define SM_WHH1   65536                     // 512*128 = 65536
#define SM_HB0    131072                    // 2*16*128 = 4096
#define SM_HB1    135168                    // 2*16*128 = 4096
#define SM_XB     139264                    // 16*64   = 1024
#define SM_HALFS  140288
#define SMEM_BYTES (SM_HALFS * 2)           // 280576 B < 320KB

union V16U { v16h v; h8 h[2]; };

// A operand (16x32 f16): lane hf=0 holds K = kb*32 + {0..7, 16..23},
// lane hf=1 holds K = kb*32 + {8..15, 24..31}. Row-major source, stride elems.
__device__ __forceinline__ v16h load_A(const _Float16* base, int stride,
                                       int row, int kb, int hf) {
  const _Float16* p = base + row * stride + kb * 32 + hf * 8;
  V16U u;
  u.h[0] = *(const h8*)(p);
  u.h[1] = *(const h8*)(p + 16);
  return u.v;
}

// B operand (32x16 f16): lane col n = lane&15; lane hf=0 holds K = kb*32 +
// {0..7, 16..23}, hf=1 holds {8..15, 24..31} from weight row n (W row-major [N][K]).
__device__ __forceinline__ v16h load_B(const _Float16* W, int K,
                                       int n, int kb, int hf) {
  const _Float16* p = W + (size_t)n * K + kb * 32 + hf * 8;
  V16U u;
  u.h[0] = *(const h8*)(p);
  u.h[1] = *(const h8*)(p + 16);
  return u.v;
}

__device__ __forceinline__ v8f wmma_f16(v16h a, v16h b, v8f c) {
  return __builtin_amdgcn_wmma_f32_16x16x32_f16(false, a, false, b,
                                                (short)0, c, false, false);
}

__device__ __forceinline__ float sigmoidf_(float x) {
  return 1.0f / (1.0f + __expf(-x));
}

// One LSTM timestep for one wave: 4 gate tiles. Recurrent path (K=128) pulls
// B tiles from LDS; input path (K=32*KXS) uses register-resident B tiles.
template <int KXS>
__device__ __forceinline__ void lstm_wave_step(
    const _Float16* hprev,            // LDS 16 x H_ row-major
    const _Float16* xin, int xStride, // LDS 16 x (32*KXS) row-major
    const _Float16* WhhLDS,           // LDS f16 [4H][H]
    const v16h* BihR,                 // [4*KXS] register B tiles, gate-major
    const float* bias4,               // per-lane fused bias, one per gate
    float* creg,                      // [8] cell state (f32)
    _Float16* hnew,                   // LDS 16 x H_ row-major
    int wave, int lane) {
  const int row = lane & 15;
  const int hf  = lane >> 4;

  v16h Ah[4];
#pragma unroll
  for (int kb = 0; kb < 4; ++kb) Ah[kb] = load_A(hprev, H_, row, kb, hf);
  v16h Ax[KXS];
#pragma unroll
  for (int kb = 0; kb < KXS; ++kb) Ax[kb] = load_A(xin, xStride, row, kb, hf);

  v8f acc[4];
#pragma unroll
  for (int g = 0; g < 4; ++g) {
    v8f a;
#pragma unroll
    for (int r = 0; r < 8; ++r) a[r] = bias4[g];
    const int n = g * H_ + wave * 16 + row;  // gate column (B-tile row in W)
#pragma unroll
    for (int kb = 0; kb < 4; ++kb)
      a = wmma_f16(Ah[kb], load_B(WhhLDS, H_, n, kb, hf), a);
#pragma unroll
    for (int kb = 0; kb < KXS; ++kb)
      a = wmma_f16(Ax[kb], BihR[g * KXS + kb], a);
    acc[g] = a;
  }

  const int mycol = wave * 16 + row;
#pragma unroll
  for (int r = 0; r < 8; ++r) {
    float ig = sigmoidf_(acc[0][r]);
    float fg = sigmoidf_(acc[1][r]);
    float gg = tanhf(acc[2][r]);
    float og = sigmoidf_(acc[3][r]);
    float c  = fg * creg[r] + ig * gg;
    creg[r]  = c;
    float hv = og * tanhf(c);
    int m = r + 8 * hf;
    hnew[m * H_ + mycol] = (_Float16)hv;
  }
}

__launch_bounds__(256, 2)
__global__ void lstm_fused(const float* __restrict__ x,
                           const float* __restrict__ h0,
                           const float* __restrict__ c0,
                           const _Float16* __restrict__ wf16,
                           const float* __restrict__ biases,  // [2][512] fused
                           const float* __restrict__ bfc,
                           float* __restrict__ out) {
  extern __shared__ __align__(16) _Float16 smem[];
  _Float16* Whh0s = smem + SM_WHH0;
  _Float16* Whh1s = smem + SM_WHH1;
  _Float16* hb0   = smem + SM_HB0;   // [2][16][H_]
  _Float16* hb1   = smem + SM_HB1;   // [2][16][H_]
  _Float16* xb    = smem + SM_XB;    // [16][IN_]

  const int tid  = threadIdx.x;
  const int wave = tid >> 5;
  const int lane = tid & 31;
  const int row  = lane & 15;
  const int hf   = lane >> 4;
  const int b0   = blockIdx.x * 16;

  // ---- prologue: W_hh0/W_hh1 (f16) -> LDS, 16B vector copies ----
  {
    const uint4* s0 = (const uint4*)(wf16 + OFF_WHH0);
    const uint4* s1 = (const uint4*)(wf16 + OFF_WHH1);
    uint4* d0 = (uint4*)(Whh0s);
    uint4* d1 = (uint4*)(Whh1s);
    for (int i = tid; i < 8192; i += 256) {  // 65536 halfs / 8 per uint4
      d0[i] = s0[i];
      d1[i] = s1[i];
    }
  }
  // ---- init h into LDS (f16) ----
  for (int i = tid; i < 16 * H_; i += 256) {
    int r = i / H_, c = i % H_;
    hb0[r * H_ + c] = (_Float16)h0[0 * B_ * H_ + (b0 + r) * H_ + c];
    hb1[r * H_ + c] = (_Float16)h0[1 * B_ * H_ + (b0 + r) * H_ + c];
  }

  // ---- per-wave register-resident weight slices ----
  const int ncol = wave * 16 + row;       // this lane's gate column offset
  v16h BihR0[4 * 2];                      // W_ih0 slice: 4 gates x K=64
  v16h BihR1[4 * 4];                      // W_ih1 slice: 4 gates x K=128
  v16h BfcR[4];                           // W_fc slice (used by waves 0..3)
#pragma unroll
  for (int g = 0; g < 4; ++g) {
#pragma unroll
    for (int kb = 0; kb < 2; ++kb)
      BihR0[g * 2 + kb] = load_B(wf16 + OFF_WIH0, IN_, g * H_ + ncol, kb, hf);
#pragma unroll
    for (int kb = 0; kb < 4; ++kb)
      BihR1[g * 4 + kb] = load_B(wf16 + OFF_WIH1, H_, g * H_ + ncol, kb, hf);
  }
  const int nfc = (wave & 3) * 16 + row;  // all waves load; waves 0..3 use
#pragma unroll
  for (int kb = 0; kb < 4; ++kb)
    BfcR[kb] = load_B(wf16 + OFF_WFC, H_, nfc, kb, hf);

  // ---- per-wave c/bias registers ----
  float c0reg[8], c1reg[8];
#pragma unroll
  for (int r = 0; r < 8; ++r) {
    int m = r + 8 * hf;
    c0reg[r] = c0[0 * B_ * H_ + (b0 + m) * H_ + ncol];
    c1reg[r] = c0[1 * B_ * H_ + (b0 + m) * H_ + ncol];
  }
  float bi0[4], bi1[4];
#pragma unroll
  for (int g = 0; g < 4; ++g) {
    bi0[g] = biases[g * H_ + ncol];          // fused layer-0 bias
    bi1[g] = biases[512 + g * H_ + ncol];    // fused layer-1 bias
  }
  const float bfc_r = bfc[nfc];
  __syncthreads();

  // ---- sequential time loop; both layers + head fused per step ----
  for (int t = 0; t < T_; ++t) {
    const int cur = (t & 1) * (16 * H_);
    const int nxt = ((t + 1) & 1) * (16 * H_);

    // stage x_t (f32 -> f16) into LDS
    for (int i = tid; i < 16 * IN_; i += 256) {
      int r = i >> 6, c = i & 63;
      xb[r * IN_ + c] = (_Float16)x[((size_t)(b0 + r) * T_ + t) * IN_ + c];
    }
    if (t + 1 < T_ && tid < 16)  // pull next step's rows toward the caches
      __builtin_prefetch(&x[((size_t)(b0 + tid) * T_ + (t + 1)) * IN_], 0, 0);
    __syncthreads();

    lstm_wave_step<2>(hb0 + cur, xb, IN_, Whh0s, BihR0,
                      bi0, c0reg, hb0 + nxt, wave, lane);
    __syncthreads();

    lstm_wave_step<4>(hb1 + cur, hb0 + nxt, H_, Whh1s, BihR1,
                      bi1, c1reg, hb1 + nxt, wave, lane);
    __syncthreads();

    // FC head: out[b, t, :] = y1_t @ Wfc^T + bfc (waves 0..3, 16-col tile)
    if (wave < 4) {
      v8f a;
#pragma unroll
      for (int r = 0; r < 8; ++r) a[r] = bfc_r;
#pragma unroll
      for (int kb = 0; kb < 4; ++kb)
        a = wmma_f16(load_A(hb1 + nxt, H_, row, kb, hf), BfcR[kb], a);
#pragma unroll
      for (int r = 0; r < 8; ++r) {
        int m = r + 8 * hf;
        out[((size_t)(b0 + m) * T_ + t) * OUT_ + nfc] = a[r];
      }
    }
    __syncthreads();
  }

  // ---- epilogue: hN [2,B,H] then cN [2,B,H] appended after output ----
  float* hN = out + (size_t)B_ * T_ * OUT_;
  float* cN = hN + 2 * B_ * H_;
  const int fin = (T_ & 1) * (16 * H_);  // last step wrote buffer 0
  for (int i = tid; i < 16 * H_; i += 256) {
    int r = i / H_, c = i % H_;
    hN[0 * B_ * H_ + (b0 + r) * H_ + c] = (float)hb0[fin + r * H_ + c];
    hN[1 * B_ * H_ + (b0 + r) * H_ + c] = (float)hb1[fin + r * H_ + c];
  }
#pragma unroll
  for (int r = 0; r < 8; ++r) {
    int m = r + 8 * hf;
    cN[0 * B_ * H_ + (b0 + m) * H_ + ncol] = c0reg[r];
    cN[1 * B_ * H_ + (b0 + m) * H_ + ncol] = c1reg[r];
  }
}

// One-shot f32 -> f16 weight conversion + bias fusion into workspace.
__global__ void convert_weights(const float* __restrict__ Wih0,
                                const float* __restrict__ Whh0,
                                const float* __restrict__ bih0,
                                const float* __restrict__ bhh0,
                                const float* __restrict__ Wih1,
                                const float* __restrict__ Whh1,
                                const float* __restrict__ bih1,
                                const float* __restrict__ bhh1,
                                const float* __restrict__ Wfc,
                                _Float16* __restrict__ wf16,
                                float* __restrict__ biases) {
  int i = blockIdx.x * 256 + threadIdx.x;
  if (i < 65536) {
    wf16[OFF_WHH0 + i] = (_Float16)Whh0[i];
  } else if (i < 98304) {
    wf16[OFF_WIH0 + (i - 65536)] = (_Float16)Wih0[i - 65536];
  } else if (i < 163840) {
    wf16[OFF_WHH1 + (i - 98304)] = (_Float16)Whh1[i - 98304];
  } else if (i < 229376) {
    wf16[OFF_WIH1 + (i - 163840)] = (_Float16)Wih1[i - 163840];
  } else if (i < 237568) {
    wf16[OFF_WFC + (i - 229376)] = (_Float16)Wfc[i - 229376];
  } else if (i < 238080) {
    int j = i - 237568;
    biases[j] = bih0[j] + bhh0[j];          // fused layer-0 bias
  } else if (i < 238592) {
    int j = i - 238080;
    biases[512 + j] = bih1[j] + bhh1[j];    // fused layer-1 bias
  }
}

extern "C" void kernel_launch(void* const* d_in, const int* in_sizes, int n_in,
                              void* d_out, int out_size, void* d_ws, size_t ws_size,
                              hipStream_t stream) {
  const float* x    = (const float*)d_in[0];
  const float* h0   = (const float*)d_in[1];
  const float* c0   = (const float*)d_in[2];
  const float* Wih0 = (const float*)d_in[3];
  const float* Whh0 = (const float*)d_in[4];
  const float* bih0 = (const float*)d_in[5];
  const float* bhh0 = (const float*)d_in[6];
  const float* Wih1 = (const float*)d_in[7];
  const float* Whh1 = (const float*)d_in[8];
  const float* bih1 = (const float*)d_in[9];
  const float* bhh1 = (const float*)d_in[10];
  const float* Wfc  = (const float*)d_in[11];
  const float* bfc  = (const float*)d_in[12];

  _Float16* wf16 = (_Float16*)d_ws;
  float* biases  = (float*)((char*)d_ws + BIAS_BYTE_OFF);

  convert_weights<<<dim3(932), dim3(256), 0, stream>>>(
      Wih0, Whh0, bih0, bhh0, Wih1, Whh1, bih1, bhh1, Wfc, wf16, biases);

  lstm_fused<<<dim3(B_ / 16), dim3(256), SMEM_BYTES, stream>>>(
      x, h0, c0, wf16, biases, bfc, (float*)d_out);
}
